// ScoreBranch_28930899706370
// MI455X (gfx1250) — matrix operand order repaired and checked by reading, not compiled, not run on hardware
//
#include <hip/hip_runtime.h>
#include <hip/hip_bf16.h>

typedef __attribute__((ext_vector_type(8)))  float  v8f;
typedef __attribute__((ext_vector_type(16))) __bf16 v16bf;

// ---------- helpers ----------
__device__ __forceinline__ unsigned short f2bf(float f) {
    unsigned u = __float_as_uint(f);
    return (unsigned short)((u + 0x7FFFu + ((u >> 16) & 1u)) >> 16);
}
__device__ __forceinline__ unsigned packbf(float a, float b) {
    return (unsigned)f2bf(a) | ((unsigned)f2bf(b) << 16);
}

// ---------- weight prepack into WMMA A-fragment order ----------
// K ordering: k = kpos*256 + ic  (kernel position outer, channel inner)
// A 16x32 bf16 lane layout (ISA 7.12.2): lanes0-15 M=lane, pairs kp = v(<4) / v+4(>=4)
//                                        lanes16-31 M=lane-16, pairs kp = v+4(<4) / v+8(>=4)
#define NWT (72*4096)   // 3x3: 72 K-blocks * 16 Mtiles * 32 lanes * 8 dwords
#define NW1 (8*4096)    // 1x1: 8 K-blocks

__device__ __forceinline__ void packA(const float* w, unsigned* dst, int KK, int idx) {
    int kb  = idx >> 12;
    int rem = idx & 4095;
    int mt  = rem >> 8;
    int L   = (rem >> 3) & 31;
    int v   = rem & 7;
    int oc  = mt * 16 + (L & 15);
    int kp;
    if (L < 16) kp = (v < 4) ? v       : (v + 4);
    else        kp = (v < 4) ? (v + 4) : (v + 8);
    int k0   = kb * 32 + kp * 2;
    int kpos = k0 >> 8;
    int ic   = k0 & 255;
    float f0 = w[(oc * 256 + ic)     * KK + kpos];
    float f1 = w[(oc * 256 + ic + 1) * KK + kpos];
    dst[idx] = packbf(f0, f1);
}

__global__ void prep_kernel(
    const float* wt, const float* ws, const float* w1,
    const float* bt, const float* gt, const float* bet, const float* mt_, const float* vt,
    const float* bs, const float* gs, const float* bes, const float* ms_, const float* vs,
    const float* b1, const float* g1, const float* be1, const float* m1_, const float* v1,
    unsigned* wtp, unsigned* wsp, unsigned* w1p, float* scsh)
{
    const int total = 2 * NWT + NW1 + 768;
    for (int gid = blockIdx.x * blockDim.x + threadIdx.x; gid < total;
         gid += gridDim.x * blockDim.x) {
        if (gid < NWT)                 packA(wt, wtp, 9, gid);
        else if (gid < 2 * NWT)        packA(ws, wsp, 9, gid - NWT);
        else if (gid < 2 * NWT + NW1)  packA(w1, w1p, 1, gid - 2 * NWT);
        else {
            int q = gid - (2 * NWT + NW1);
            int which = q >> 8, c = q & 255;
            const float *g, *bi, *be, *m, *v;
            if (which == 0)      { g = gt; bi = bt; be = bet; m = mt_; v = vt; }
            else if (which == 1) { g = gs; bi = bs; be = bes; m = ms_; v = vs; }
            else                 { g = g1; bi = b1; be = be1; m = m1_; v = v1; }
            float inv = g[c] * rsqrtf(v[c] + 1e-5f);
            scsh[q * 2]     = inv;                          // scale
            scsh[q * 2 + 1] = (bi[c] - m[c]) * inv + be[c]; // conv-bias + BN shift
        }
    }
}

// ---------- fused conv(3x3 or 1x1) + BN + ReLU as WMMA implicit GEMM ----------
// WG = 256 threads = 8 waves. WG tile: 256 oc x 32 pixels (2 N-tiles).
// Wave: Mtiles {w, w+8} x Ntiles {0,1} -> 4 WMMAs per K-step; each A fragment
// feeds 2 WMMAs (halves L2 weight traffic vs a 16-pixel tile).
// B fragments staged in LDS fragment-ready: d = kb*512 + half*256 + lane*8 + v
__global__ void __launch_bounds__(256) conv_wmma(
    const float* __restrict__ in, const unsigned* __restrict__ apack,
    const float* __restrict__ scsh, float* __restrict__ out,
    int Hin, int Win, int Hout, int Wout, int KW, int KB)
{
    extern __shared__ unsigned ldsB[];
    const int tid  = threadIdx.x;
    const int tile = blockIdx.x;
    const int P    = Hout * Wout;
    const int ICS  = Hin * Win;

    // --- stage im2col B tile (32 pixels x KB*32 K) into LDS as bf16 pairs ---
    for (int d = tid; d < KB * 512; d += 256) {
        int kb = d >> 9, r = d & 511;
        int half = r >> 8, r8 = r & 255;
        int L = r8 >> 3, v = r8 & 7;
        int n  = half * 16 + (L & 15);
        int kp = (L < 16) ? ((v < 4) ? v : (v + 4)) : ((v < 4) ? (v + 4) : (v + 8));          // B 32x16 lane layout
        int k0 = kb * 32 + kp * 2;
        int kpos = k0 >> 8, ic = k0 & 255;
        int kh = kpos / KW, kw = kpos % KW;
        int pix = tile * 32 + n;
        int b = pix / P, rr = pix - b * P;
        int oy = rr / Wout, ox = rr - oy * Wout;
        int base = ((b * 256 + ic) * Hin + (oy + kh)) * Win + (ox + kw);
        ldsB[d] = packbf(in[base], in[base + ICS]);
    }
    __syncthreads();

    const int lane = tid & 31, wv = tid >> 5;
    const int mt0 = wv, mt1 = wv + 8;
    const unsigned* ap0 = apack + ((size_t)mt0 * 256 + lane * 8);
    const unsigned* ap1 = apack + ((size_t)mt1 * 256 + lane * 8);
    v8f c00 = {}, c01 = {}, c10 = {}, c11 = {};
    for (int kb = 0; kb < KB; ++kb) {
        // prefetch next K-step's A fragments into cache (global_prefetch_b8)
        if (kb + 1 < KB) {
            __builtin_prefetch(ap0 + (kb + 1) * 4096, 0, 3);
            __builtin_prefetch(ap1 + (kb + 1) * 4096, 0, 3);
        }
        v16bf B0 = *(const v16bf*)&ldsB[kb * 512 + lane * 8];
        v16bf B1 = *(const v16bf*)&ldsB[kb * 512 + 256 + lane * 8];
        v16bf A0 = *(const v16bf*)(ap0 + kb * 4096);
        v16bf A1 = *(const v16bf*)(ap1 + kb * 4096);
        c00 = __builtin_amdgcn_wmma_f32_16x16x32_bf16(false, A0, false, B0,
                                                      (short)0, c00, false, false);
        c01 = __builtin_amdgcn_wmma_f32_16x16x32_bf16(false, A0, false, B1,
                                                      (short)0, c01, false, false);
        c10 = __builtin_amdgcn_wmma_f32_16x16x32_bf16(false, A1, false, B0,
                                                      (short)0, c10, false, false);
        c11 = __builtin_amdgcn_wmma_f32_16x16x32_bf16(false, A1, false, B1,
                                                      (short)0, c11, false, false);
    }

    // --- epilogue: scale/shift (bias+BN) + ReLU, scatter to NCHW ---
    const int n  = lane & 15;
    const int hi = (lane >> 4) << 3;              // C/D layout: M = r + 8*(lane>=16)
#pragma unroll
    for (int half = 0; half < 2; ++half) {
        int pix = tile * 32 + half * 16 + n;
        int b = pix / P, rr = pix - b * P;
        int oy = rr / Wout, ox = rr - oy * Wout;
        const v8f* cA = half ? &c01 : &c00;
        const v8f* cB = half ? &c11 : &c10;
#pragma unroll
        for (int r = 0; r < 8; ++r) {
            int oc0 = mt0 * 16 + hi + r;
            float v0 = fmaxf((*cA)[r] * scsh[oc0 * 2] + scsh[oc0 * 2 + 1], 0.f);
            out[((b * 256 + oc0) * Hout + oy) * Wout + ox] = v0;
            int oc1 = mt1 * 16 + hi + r;
            float v1 = fmaxf((*cB)[r] * scsh[oc1 * 2] + scsh[oc1 * 2 + 1], 0.f);
            out[((b * 256 + oc1) * Hout + oy) * Wout + ox] = v1;
        }
    }
}

// ---------- depthwise cross-correlation (per (b,c) channel) ----------
template <int T, int PAD>
__global__ void __launch_bounds__(128) corr_kernel(
    const float* __restrict__ tmpl, const float* __restrict__ srch,
    float* __restrict__ out, int O)
{
    __shared__ float st[49];
    __shared__ float sS[29 * 29];
    const int bc  = blockIdx.x;                   // b*256 + c
    const int tid = threadIdx.x;
    const float* tp = tmpl + bc * T * T;
    const float* sp = srch + bc * 29 * 29;
    for (int i = tid; i < T * T;   i += 128) st[i] = tp[i];
    for (int i = tid; i < 29 * 29; i += 128) sS[i] = sp[i];
    __syncthreads();
    const int OO = O * O;
    for (int p = tid; p < OO; p += 128) {
        int oy = p / O, ox = p - oy * O;
        float acc = 0.f;
#pragma unroll
        for (int i = 0; i < T; ++i)
#pragma unroll
            for (int j = 0; j < T; ++j) {
                int iy = oy + i - PAD, ix = ox + j - PAD;
                if (PAD == 0 || ((unsigned)iy < 29u && (unsigned)ix < 29u))
                    acc += st[i * T + j] * sS[iy * 29 + ix];
            }
        out[bc * OO + p] = acc;
    }
}

// ---------- final 256->1 1x1 conv + sigmoid ----------
__global__ void __launch_bounds__(256) final_kernel(
    const float* __restrict__ h1, const float* __restrict__ w2,
    const float* __restrict__ b2, float* __restrict__ out, int P)
{
    __shared__ float w[256];
    const int tid = threadIdx.x;
    w[tid] = w2[tid];
    __syncthreads();
    int idx = blockIdx.x * 256 + tid;
    if (idx < 64 * P) {
        int b = idx / P, p = idx - b * P;
        const float* hp = h1 + (size_t)(b * 256) * P + p;
        float acc = b2[0];
        for (int ic = 0; ic < 256; ++ic) acc += w[ic] * hp[ic * P];
        out[idx] = 1.f / (1.f + __expf(-acc));
    }
}

// ---------- host orchestration ----------
extern "C" void kernel_launch(void* const* d_in, const int* in_sizes, int n_in,
                              void* d_out, int out_size, void* d_ws, size_t ws_size,
                              hipStream_t stream) {
    (void)in_sizes; (void)n_in; (void)out_size; (void)ws_size;
    const float* large  = (const float*)d_in[0];
    const float* medium = (const float*)d_in[1];
    const float* small  = (const float*)d_in[2];
    const float* search = (const float*)d_in[3];
    const float* wt  = (const float*)d_in[4];
    const float* bt  = (const float*)d_in[5];
    const float* gt  = (const float*)d_in[6];
    const float* bet = (const float*)d_in[7];
    const float* mt_ = (const float*)d_in[8];
    const float* vt  = (const float*)d_in[9];
    const float* ws2 = (const float*)d_in[10];
    const float* bs  = (const float*)d_in[11];
    const float* gs  = (const float*)d_in[12];
    const float* bes = (const float*)d_in[13];
    const float* ms_ = (const float*)d_in[14];
    const float* vs  = (const float*)d_in[15];
    const float* w1  = (const float*)d_in[16];
    const float* b1  = (const float*)d_in[17];
    const float* g1  = (const float*)d_in[18];
    const float* be1 = (const float*)d_in[19];
    const float* m1_ = (const float*)d_in[20];
    const float* v1  = (const float*)d_in[21];
    const float* w2  = (const float*)d_in[22];
    const float* b2  = (const float*)d_in[23];

    char* ws = (char*)d_ws;
    unsigned* wtp  = (unsigned*)(ws + (size_t)0);
    unsigned* wsp  = (unsigned*)(ws + ((size_t)2  << 20));
    unsigned* w1p  = (unsigned*)(ws + ((size_t)4  << 20));
    float*    scsh = (float*)   (ws + ((size_t)5  << 20));   // 3 x 256 x {scale,shift}
    float*    lf   = (float*)   (ws + ((size_t)6  << 20));   // (64,256,7,7)
    float*    mf   = (float*)   (ws + ((size_t)10 << 20));   // (64,256,5,5)
    float*    sf   = (float*)   (ws + ((size_t)12 << 20));   // (64,256,3,3)
    float*    sout = (float*)   (ws + ((size_t)16 << 20));   // (64,256,29,29)
    float*    corr = (float*)   (ws + ((size_t)80 << 20));   // reused per branch
    float*    h1   = (float*)   (ws + ((size_t)128<< 20));   // reused per branch

    prep_kernel<<<2436, 256, 0, stream>>>(wt, ws2, w1,
        bt, gt, bet, mt_, vt, bs, gs, bes, ms_, vs, b1, g1, be1, m1_, v1,
        wtp, wsp, w1p, scsh);

    // conv_target on three template scales (shared weights), conv_search
    conv_wmma<<<98,   256, 72*2048, stream>>>(large,  wtp, scsh,        lf,   9,  9,  7,  7, 3, 72);
    conv_wmma<<<50,   256, 72*2048, stream>>>(medium, wtp, scsh,        mf,   7,  7,  5,  5, 3, 72);
    conv_wmma<<<18,   256, 72*2048, stream>>>(small,  wtp, scsh,        sf,   5,  5,  3,  3, 3, 72);
    conv_wmma<<<1682, 256, 72*2048, stream>>>(search, wsp, scsh + 512,  sout, 31, 31, 29, 29, 3, 72);

    float* out = (float*)d_out;

    // branch L: 7x7 template, pad 2 -> 27x27
    corr_kernel<7,2><<<16384, 128, 0, stream>>>(lf, sout, corr, 27);
    conv_wmma<<<1458, 256, 8*2048, stream>>>(corr, w1p, scsh + 1024, h1, 27, 27, 27, 27, 1, 8);
    final_kernel<<<(64*729 + 255)/256, 256, 0, stream>>>(h1, w2, b2, out, 729);

    // branch M: 5x5 template, pad 0 -> 25x25
    corr_kernel<5,0><<<16384, 128, 0, stream>>>(mf, sout, corr, 25);
    conv_wmma<<<1250, 256, 8*2048, stream>>>(corr, w1p, scsh + 1024, h1, 25, 25, 25, 25, 1, 8);
    final_kernel<<<(64*625 + 255)/256, 256, 0, stream>>>(h1, w2, b2, out + 46656, 625);

    // branch S: 3x3 template, pad 0 -> 27x27
    corr_kernel<3,0><<<16384, 128, 0, stream>>>(sf, sout, corr, 27);
    conv_wmma<<<1458, 256, 8*2048, stream>>>(corr, w1p, scsh + 1024, h1, 27, 27, 27, 27, 1, 8);
    final_kernel<<<(64*729 + 255)/256, 256, 0, stream>>>(h1, w2, b2, out + 86656, 729);
}
